// FlowNet3DFeatureCat4l_1872605741583
// MI455X (gfx1250) — matrix operand order repaired and checked by reading, not compiled, not run on hardware
//
#include <hip/hip_runtime.h>
#include <hip/hip_bf16.h>

#define BB 4
#define NN 16384

typedef __attribute__((ext_vector_type(16))) _Float16 v16h;
typedef __attribute__((ext_vector_type(8)))  _Float16 v8h;
typedef __attribute__((ext_vector_type(8)))  float    v8f;

// ---------------------------------------------------------------------------
// pc2 (B,3,N) -> xyz (B,N,3)
// ---------------------------------------------------------------------------
__global__ void k_transpose(const float* __restrict__ pc, float* __restrict__ xyz)
{
    size_t g = (size_t)blockIdx.x * blockDim.x + threadIdx.x;
    if (g >= (size_t)BB * NN * 3) return;
    int    c  = (int)(g % 3);
    size_t bn = g / 3;
    int    n  = (int)(bn % NN);
    int    b  = (int)(bn / NN);
    xyz[g] = pc[((size_t)b * 3 + c) * NN + n];
}

// ---------------------------------------------------------------------------
// Farthest point sampling: one block per batch, dist[] lives in LDS.
// 1024 serial argmax rounds -> block tree reduction via ds ops.
// ---------------------------------------------------------------------------
__global__ void k_fps(const float* __restrict__ xyz, int Nin, int npoint,
                      int* __restrict__ fidx)
{
    extern __shared__ float smem[];
    float* dist = smem;                       // Nin floats
    float* rv   = smem + Nin;                 // blockDim floats
    int*   ri   = (int*)(rv + blockDim.x);    // blockDim ints

    const int b = blockIdx.x;
    const float* p = xyz + (size_t)b * Nin * 3;

    for (int i = threadIdx.x; i < Nin; i += blockDim.x) dist[i] = 1e10f;
    __syncthreads();

    int far = 0;
    for (int t = 0; t < npoint; ++t) {
        if (threadIdx.x == 0) fidx[b * npoint + t] = far;
        float cx = p[far * 3 + 0], cy = p[far * 3 + 1], cz = p[far * 3 + 2];
        float bestv = -1.0f; int besti = 0;
        for (int i = threadIdx.x; i < Nin; i += blockDim.x) {
            float dx = p[i * 3 + 0] - cx;
            float dy = p[i * 3 + 1] - cy;
            float dz = p[i * 3 + 2] - cz;
            float d  = dx * dx + dy * dy + dz * dz;
            float dd = fminf(dist[i], d);
            dist[i] = dd;
            if (dd > bestv) { bestv = dd; besti = i; }
        }
        rv[threadIdx.x] = bestv; ri[threadIdx.x] = besti;
        __syncthreads();
        for (int s = blockDim.x >> 1; s > 0; s >>= 1) {
            if (threadIdx.x < s) {
                float v2 = rv[threadIdx.x + s]; int i2 = ri[threadIdx.x + s];
                if (v2 > rv[threadIdx.x] ||
                    (v2 == rv[threadIdx.x] && i2 < ri[threadIdx.x])) {
                    rv[threadIdx.x] = v2; ri[threadIdx.x] = i2;
                }
            }
            __syncthreads();
        }
        far = ri[0];
        __syncthreads();
    }
}

// ---------------------------------------------------------------------------
// Ball query: sort(idx)[:K] == first K indices (ascending) within radius.
// Centroid is one of the points (d=0) so a pad index always exists.
// Also emits new_xyz.
// ---------------------------------------------------------------------------
__global__ void k_ballq(const float* __restrict__ xyz, int Nin,
                        const int* __restrict__ fidx, int S, int K, float r2,
                        float* __restrict__ new_xyz, int* __restrict__ idx)
{
    int g = blockIdx.x * blockDim.x + threadIdx.x;
    if (g >= BB * S) return;
    int b = g / S;
    const float* p = xyz + (size_t)b * Nin * 3;
    int fi = fidx[g];
    float cx = p[fi * 3 + 0], cy = p[fi * 3 + 1], cz = p[fi * 3 + 2];
    new_xyz[(size_t)g * 3 + 0] = cx;
    new_xyz[(size_t)g * 3 + 1] = cy;
    new_xyz[(size_t)g * 3 + 2] = cz;
    int* out = idx + (size_t)g * K;
    int cnt = 0, first = fi;
    for (int i = 0; i < Nin && cnt < K; ++i) {
        float dx = p[i * 3 + 0] - cx;
        float dy = p[i * 3 + 1] - cy;
        float dz = p[i * 3 + 2] - cz;
        if (dx * dx + dy * dy + dz * dz <= r2) {
            if (cnt == 0) first = i;
            out[cnt++] = i;
        }
    }
    for (; cnt < K; ++cnt) out[cnt] = first;
}

// ---------------------------------------------------------------------------
// Gather + concat [rel_xyz | point feats | zero pad] -> f16 A (B*S*K, Kpad)
// ---------------------------------------------------------------------------
__global__ void k_group(const float* __restrict__ xyz,
                        const float* __restrict__ pts, int Nin, int Cpts,
                        const float* __restrict__ new_xyz,
                        const int* __restrict__ idx, int S, int K, int Kpad,
                        _Float16* __restrict__ A)
{
    size_t g = (size_t)blockIdx.x * blockDim.x + threadIdx.x;
    size_t total = (size_t)BB * S * K * Kpad;
    if (g >= total) return;
    int    c   = (int)(g % Kpad);
    size_t row = g / Kpad;
    int    k   = (int)(row % K);
    size_t bs  = row / K;              // b*S + s
    size_t b   = bs / S;
    float v = 0.0f;
    if (c < 3 + Cpts) {
        int id = idx[bs * K + k];
        if (c < 3)
            v = xyz[(b * Nin + id) * 3 + c] - new_xyz[bs * 3 + c];
        else
            v = pts[(b * Nin + id) * (size_t)Cpts + (c - 3)];
    }
    A[g] = (_Float16)v;
}

// ---------------------------------------------------------------------------
// W f32 (Cout,Cin) -> f16 (Cout,Kpad), zero padded in K
// ---------------------------------------------------------------------------
__global__ void k_prep_w(const float* __restrict__ W, int Cout, int Cin,
                         int Kpad, _Float16* __restrict__ Wt)
{
    size_t g = (size_t)blockIdx.x * blockDim.x + threadIdx.x;
    if (g >= (size_t)Cout * Kpad) return;
    int c = (int)(g % Kpad);
    size_t o = g / Kpad;
    Wt[g] = (c < Cin) ? (_Float16)W[o * Cin + c] : (_Float16)0.0f;
}

// ---------------------------------------------------------------------------
// WMMA GEMM: Y = epilogue( A(M,Kp) @ Wt(Cout,Kp)^T ).
// One wave per 16x16 tile; v_wmma_f32_16x16x32_f16 K-loop.
// A frag (16-bit A 16x32): lane m=L&15, 16 contiguous K = kb+{0..15}, same order as B.
// B frag: lane n=L&15, 16 contiguous K starting at (L>=16 ? 16 : 0).
// relu_bn==1 : y = relu((acc + b - m)*g*rsqrt(v+eps) + be)
// relu_bn==0 : y = acc + b
// ---------------------------------------------------------------------------
__global__ void __launch_bounds__(128)
k_wmma_gemm(const _Float16* __restrict__ A, const _Float16* __restrict__ Wt,
            const float* __restrict__ bias, const float* __restrict__ gamma,
            const float* __restrict__ beta, const float* __restrict__ mean,
            const float* __restrict__ var,
            float* __restrict__ Yf32, _Float16* __restrict__ Yf16,
            int M, int Kp, int Cout, int relu_bn)
{
    const int lane = threadIdx.x & 31;
    const int wave = blockIdx.x * (blockDim.x >> 5) + (threadIdx.x >> 5);
    const int tilesN = Cout >> 4;
    const int tilesM = M >> 4;
    if (wave >= tilesM * tilesN) return;          // wave-uniform: EXEC stays full
    const int tm = wave / tilesN;
    const int tn = wave % tilesN;

    const int mrow = tm * 16 + (lane & 15);
    const int kb   = (lane >> 4) * 16;            // A half-wave K base
    const int ocol = tn * 16 + (lane & 15);
    const int koff = (lane >> 4) * 16;            // B half-wave K base

    const _Float16* arow = A  + (size_t)mrow * Kp + kb;
    const _Float16* brow = Wt + (size_t)ocol * Kp + koff;

    v8f acc = {0.f, 0.f, 0.f, 0.f, 0.f, 0.f, 0.f, 0.f};
    for (int kc = 0; kc < Kp; kc += 32) {
        __builtin_prefetch((const void*)(brow + kc + 128), 0, 1);
        v8h alo = *(const v8h*)(arow + kc);
        v8h ahi = *(const v8h*)(arow + kc + 8);
        v16h afrag;
#pragma unroll
        for (int i = 0; i < 8; ++i) { afrag[i] = alo[i]; afrag[i + 8] = ahi[i]; }
        v16h bfrag = *(const v16h*)(brow + kc);
        acc = __builtin_amdgcn_wmma_f32_16x16x32_f16(
                  false, afrag, false, bfrag, (short)0, acc, false, false);
    }

    float sc = 1.0f, sh = bias[ocol];
    if (relu_bn) {
        sc = gamma[ocol] * rsqrtf(var[ocol] + 1e-5f);
        sh = (bias[ocol] - mean[ocol]) * sc + beta[ocol];
    }
    const int rbase = tm * 16 + ((lane >> 4) << 3);
#pragma unroll
    for (int r = 0; r < 8; ++r) {
        float y = acc[r] * sc + sh;
        if (relu_bn) y = fmaxf(y, 0.0f);
        size_t o = (size_t)(rbase + r) * Cout + ocol;
        Yf32[o] = y;
        Yf16[o] = (_Float16)y;
    }
}

// ---------------------------------------------------------------------------
// max over neighborhood K: (B*S*K, C) -> (B*S, C)
// ---------------------------------------------------------------------------
__global__ void k_maxpool_k(const float* __restrict__ Y, int S, int K, int C,
                            float* __restrict__ out)
{
    size_t g = (size_t)blockIdx.x * blockDim.x + threadIdx.x;
    if (g >= (size_t)BB * S * C) return;
    int c = (int)(g % C);
    size_t bs = g / C;
    const float* base = Y + (bs * K) * (size_t)C + c;
    float m = base[0];
    for (int k = 1; k < K; ++k) m = fmaxf(m, base[(size_t)k * C]);
    out[bs * (size_t)C + c] = m;
}

// ---------------------------------------------------------------------------
// max + mean over S: (B,S,C) -> (B,C) each
// ---------------------------------------------------------------------------
__global__ void k_pool_s(const float* __restrict__ in, int S, int C,
                         float* __restrict__ omax, float* __restrict__ omean)
{
    int g = blockIdx.x * blockDim.x + threadIdx.x;
    if (g >= BB * C) return;
    int b = g / C, c = g % C;
    const float* p = in + ((size_t)b * S) * C + c;
    float m = p[0], sum = p[0];
    for (int s = 1; s < S; ++s) {
        float v = p[(size_t)s * C];
        m = fmaxf(m, v); sum += v;
    }
    omax[g]  = m;
    omean[g] = sum / (float)S;
}

// ---------------------------------------------------------------------------
// Build head input (16,Kp) f16: rows 0..3 = [s0[b] | s1[b]], rows 4..15 zero
// ---------------------------------------------------------------------------
__global__ void k_seq_in(const float* __restrict__ s0, int C0,
                         const float* __restrict__ s1, int C1, int Kp,
                         _Float16* __restrict__ A)
{
    int g = blockIdx.x * blockDim.x + threadIdx.x;
    if (g >= 16 * Kp) return;
    int c = g % Kp, r = g / Kp;
    float v = 0.0f;
    if (r < BB) {
        if (c < C0)            v = s0[(size_t)r * C0 + c];
        else if (c < C0 + C1)  v = s1[(size_t)r * C1 + (c - C0)];
    }
    A[g] = (_Float16)v;
}

// ---------------------------------------------------------------------------
// copy rows 0..3 of (16,C) f32 into (4,C)
// ---------------------------------------------------------------------------
__global__ void k_copy4(const float* __restrict__ Y, int C, float* __restrict__ dst)
{
    int g = blockIdx.x * blockDim.x + threadIdx.x;
    if (g >= BB * C) return;
    dst[g] = Y[(size_t)(g / C) * C + (g % C)];
}

// ===========================================================================
extern "C" void kernel_launch(void* const* d_in, const int* in_sizes, int n_in,
                              void* d_out, int out_size, void* d_ws, size_t ws_size,
                              hipStream_t stream)
{
    (void)in_sizes; (void)n_in; (void)out_size; (void)ws_size;

    const float* pc2 = (const float*)d_in[1];
    float* dout = (float*)d_out;

    // params flattened: dict keys sorted (sa1..sa4, seq1..seq4); per-layer keys
    // sorted: W, b, be, g, m, v.  Global layer j -> d_in[2 + j*6 + t].
    auto P = [&](int j, int t) { return (const float*)d_in[2 + j * 6 + t]; };

    // -------- workspace carve-out (deterministic) --------
    char* w = (char*)d_ws;
    auto alloc = [&](size_t bytes) {
        char* p = w; w += (bytes + 255) & ~(size_t)255; return (void*)p;
    };
    float*     xyz  = (float*)alloc((size_t)BB * NN * 3 * 4);
    int*       fidx = (int*)  alloc((size_t)BB * 1024 * 4);
    int*       idxb = (int*)  alloc((size_t)BB * 1024 * 16 * 4);
    float*     l1x  = (float*)alloc((size_t)BB * 1024 * 3 * 4);
    float*     l2x  = (float*)alloc((size_t)BB * 256 * 3 * 4);
    float*     l3x  = (float*)alloc((size_t)BB * 64 * 3 * 4);
    float*     l4x  = (float*)alloc((size_t)BB * 16 * 3 * 4);
    float*     l1f  = (float*)alloc((size_t)BB * 1024 * 64 * 4);
    float*     l2f  = (float*)alloc((size_t)BB * 256 * 256 * 4);
    float*     l3f  = (float*)alloc((size_t)BB * 64 * 1024 * 4);
    float*     l4f  = (float*)alloc((size_t)BB * 16 * 2048 * 4);
    float*     g1   = (float*)alloc((size_t)BB * 64 * 4);
    float*     g2   = (float*)alloc((size_t)BB * 256 * 4);
    float*     g3   = (float*)alloc((size_t)BB * 1024 * 4);
    float*     g4   = (float*)alloc((size_t)BB * 2048 * 4);
    _Float16*  Abuf = (_Float16*)alloc((size_t)2097152 * 2);       // max A rows*Kpad
    _Float16*  Wtb  = (_Float16*)alloc((size_t)4096 * 4096 * 2);   // max weight f16
    float*     Y0f  = (float*)   alloc((size_t)4194304 * 4);
    _Float16*  Y0h  = (_Float16*)alloc((size_t)4194304 * 2);
    float*     Y1f  = (float*)   alloc((size_t)4194304 * 4);
    _Float16*  Y1h  = (_Float16*)alloc((size_t)4194304 * 2);

    auto nb = [](size_t n) { return (unsigned)((n + 255) / 256); };
    auto pad32 = [](int x) { return (x + 31) & ~31; };

    // W prep + WMMA GEMM (+ fused BN/ReLU epilogue)
    auto gemm = [&](const _Float16* Ain, int M, int Kp, int lyr, int Cin,
                    int Cout, float* Yf, _Float16* Yh, int relu_bn) {
        k_prep_w<<<nb((size_t)Cout * Kp), 256, 0, stream>>>(P(lyr, 0), Cout, Cin, Kp, Wtb);
        int waves = (M >> 4) * (Cout >> 4);
        k_wmma_gemm<<<(waves + 3) / 4, 128, 0, stream>>>(
            Ain, Wtb, P(lyr, 1), P(lyr, 3), P(lyr, 2), P(lyr, 4), P(lyr, 5),
            Yf, Yh, M, Kp, Cout, relu_bn);
    };

    // One set-abstraction stage
    auto run_sa = [&](const float* xin, int Nin, const float* pts, int Cpts,
                      int S, int K, float rad, int lyr0,
                      int C1, int C2, int C3, float* nxout, float* pooled) {
        k_fps<<<BB, 256, (size_t)Nin * 4 + 256 * 8, stream>>>(xin, Nin, S, fidx);
        k_ballq<<<nb((size_t)BB * S), 256, 0, stream>>>(
            xin, Nin, fidx, S, K, rad * rad, nxout, idxb);
        int Kp0 = pad32(3 + Cpts);
        int M   = BB * S * K;
        k_group<<<nb((size_t)M * Kp0), 256, 0, stream>>>(
            xin, pts, Nin, Cpts, nxout, idxb, S, K, Kp0, Abuf);
        gemm(Abuf, M, Kp0, lyr0 + 0, 3 + Cpts, C1, Y0f, Y0h, 1);
        gemm(Y0h,  M, C1,  lyr0 + 1, C1,      C2, Y1f, Y1h, 1);
        gemm(Y1h,  M, C2,  lyr0 + 2, C2,      C3, Y0f, Y0h, 1);
        k_maxpool_k<<<nb((size_t)BB * S * C3), 256, 0, stream>>>(Y0f, S, K, C3, pooled);
    };

    // -------- pipeline --------
    k_transpose<<<nb((size_t)BB * NN * 3), 256, 0, stream>>>(pc2, xyz);

    run_sa(xyz, NN,   xyz, 3,    1024, 16, 0.5f, 0,  32,   32,   64,   l1x, l1f);
    run_sa(l1x, 1024, l1f, 64,   256,  16, 1.0f, 3,  64,   64,   256,  l2x, l2f);
    run_sa(l2x, 256,  l2f, 256,  64,   8,  2.0f, 6,  256,  256,  1024, l3x, l3f);
    run_sa(l3x, 64,   l3f, 1024, 16,   8,  4.0f, 9,  1024, 1024, 2048, l4x, l4f);

    // d_out layout (f32): out | f3 | f2 | f1 | mean4 | mean3 | mean2 | mean1
    float* OUT = dout;
    float* F3  = dout + 16384;
    float* F2  = dout + 24576;
    float* F1  = dout + 28672;
    float* M4  = dout + 29696;
    float* M3  = dout + 37888;
    float* M2  = dout + 41984;
    float* M1  = dout + 43008;

    k_pool_s<<<nb(BB * 64),   256, 0, stream>>>(l1f, 1024, 64,   g1, M1);
    k_pool_s<<<nb(BB * 256),  256, 0, stream>>>(l2f, 256,  256,  g2, M2);
    k_pool_s<<<nb(BB * 1024), 256, 0, stream>>>(l3f, 64,   1024, g3, M3);
    k_pool_s<<<nb(BB * 2048), 256, 0, stream>>>(l4f, 16,   2048, g4, M4);

    // seq1: 64 -> 128 -> 256
    k_seq_in<<<nb(16 * 64), 256, 0, stream>>>(g1, 64, nullptr, 0, 64, Abuf);
    gemm(Abuf, 16, 64, 12, 64, 128, Y0f, Y0h, 1);
    gemm(Y0h, 16, 128, 13, 128, 256, Y1f, Y1h, 1);
    k_copy4<<<nb(BB * 256), 256, 0, stream>>>(Y1f, 256, F1);

    // seq2: [256|256] -> 512 -> 1024
    k_seq_in<<<nb(16 * 512), 256, 0, stream>>>(g2, 256, F1, 256, 512, Abuf);
    gemm(Abuf, 16, 512, 14, 512, 512, Y0f, Y0h, 1);
    gemm(Y0h, 16, 512, 15, 512, 1024, Y1f, Y1h, 1);
    k_copy4<<<nb(BB * 1024), 256, 0, stream>>>(Y1f, 1024, F2);

    // seq3: [1024|1024] -> 2048 -> 2048
    k_seq_in<<<nb(16 * 2048), 256, 0, stream>>>(g3, 1024, F2, 1024, 2048, Abuf);
    gemm(Abuf, 16, 2048, 16, 2048, 2048, Y0f, Y0h, 1);
    gemm(Y0h, 16, 2048, 17, 2048, 2048, Y1f, Y1h, 1);
    k_copy4<<<nb(BB * 2048), 256, 0, stream>>>(Y1f, 2048, F3);

    // seq4: [2048|2048] -> 4096 -> 4096 (last layer: bias only, no BN/ReLU)
    k_seq_in<<<nb(16 * 4096), 256, 0, stream>>>(g4, 2048, F3, 2048, 4096, Abuf);
    gemm(Abuf, 16, 4096, 18, 4096, 4096, Y0f, Y0h, 1);
    gemm(Y0h, 16, 4096, 19, 4096, 4096, Y1f, Y1h, 0);
    k_copy4<<<nb(BB * 4096), 256, 0, stream>>>(Y1f, 4096, OUT);
}
